// LongformerBlock_68547678044786
// MI455X (gfx1250) — matrix operand order repaired and checked by reading, not compiled, not run on hardware
//
#include <hip/hip_runtime.h>
#include <hip/hip_bf16.h>
#include <math.h>

typedef __attribute__((ext_vector_type(16))) _Float16 v16h;
typedef __attribute__((ext_vector_type(8)))  _Float16 v8h;
typedef __attribute__((ext_vector_type(8)))  float    v8f;

#define D_MODEL 1024
#define TSEQ    2048
#define BATCH   2
#define NH      16
#define HD      64
#define MROWS   (BATCH*TSEQ)   /* 4096 */
#define DFF     (4*D_MODEL)    /* 4096 */

/* ---------------- GEMM config: 128x128 block, BK=64, 8 waves (4x2), wave tile 32x64 ---- */
#define BM 128
#define BN 128
#define BKK 64
#define LDS_STRIDE (BKK + 8)   /* +16B pad: conflict-free 16-lane b128 frag reads (144B = 9*16) */

__device__ __forceinline__ v16h cat8(v8h lo, v8h hi) {
  return __builtin_shufflevector(lo, hi, 0,1,2,3,4,5,6,7,8,9,10,11,12,13,14,15);
}

/* ---- CDNA5 async global->LDS staging (ASYNCcnt path, no VGPR round-trip) ---- */
__device__ __forceinline__ void async_stage64B(const void* sbase, unsigned goff, unsigned ldsoff) {
  /* copy 64 contiguous bytes: 4x b128, INST_OFFSET advances both LDS and global address */
  asm volatile(
      "global_load_async_to_lds_b128 %0, %1, %2 offset:0\n\t"
      "global_load_async_to_lds_b128 %0, %1, %2 offset:16\n\t"
      "global_load_async_to_lds_b128 %0, %1, %2 offset:32\n\t"
      "global_load_async_to_lds_b128 %0, %1, %2 offset:48"
      :: "v"(ldsoff), "v"(goff), "s"(sbase) : "memory");
}
__device__ __forceinline__ void wait_async0() {
  asm volatile("s_wait_asynccnt 0x0" ::: "memory");
}

/* Core tiled GEMM: C(2x4 tiles of 16x16 f32) += A[M,K] x Bt[N,K]^T, f16 in, f32 accum.
   Double-buffered LDS, async global->LDS staging overlapped with WMMA compute. */
__device__ __forceinline__ void gemm_core(const _Float16* __restrict__ A,
                                          const _Float16* __restrict__ Bt,
                                          int K,
                                          _Float16 (&As)[2][BM * LDS_STRIDE],
                                          _Float16 (&Bs)[2][BN * LDS_STRIDE],
                                          int bm, int bn, v8f (&acc)[2][4]) {
  const int tid  = threadIdx.x;
  const int lane = tid & 31;
  const int wid  = tid >> 5;
  const int wm   = wid & 3;        /* 4 waves along M */
  const int wn   = wid >> 2;       /* 2 waves along N */

  /* each thread stages 64B of A and 64B of B per buffer fill */
  const int r  = tid >> 1;
  const int c0 = (tid & 1) * 32;
  unsigned aoff[2], boff[2];
#pragma unroll
  for (int p = 0; p < 2; ++p) {
    aoff[p] = (unsigned)(size_t)(&As[p][r * LDS_STRIDE + c0]);  /* low 32b = LDS offset */
    boff[p] = (unsigned)(size_t)(&Bs[p][r * LDS_STRIDE + c0]);
  }
  const unsigned gA0 = (unsigned)(((size_t)(bm + r) * K + c0) * 2);
  const unsigned gB0 = (unsigned)(((size_t)(bn + r) * K + c0) * 2);
  const int nk = K / BKK;

  /* prologue: fill buffer 0 */
  async_stage64B(A,  gA0, aoff[0]);
  async_stage64B(Bt, gB0, boff[0]);

  for (int it = 0; it < nk; ++it) {
    wait_async0();          /* my async writes (stage it) done          */
    __syncthreads();        /* everyone's writes done, prior reads done */

    if (it + 1 < nk) {      /* overlap: issue stage it+1 into other buffer */
      const unsigned gk = (unsigned)((it + 1) * BKK * 2);
      async_stage64B(A,  gA0 + gk, aoff[(it + 1) & 1]);
      async_stage64B(Bt, gB0 + gk, boff[(it + 1) & 1]);
    }

    const _Float16* Asc = As[it & 1];
    const _Float16* Bsc = Bs[it & 1];
#pragma unroll
    for (int ks = 0; ks < 2; ++ks) {           /* two K=32 WMMA substeps per stage */
      const int koff = ks * 32;
      v16h af[2], bf[4];
#pragma unroll
      for (int mi = 0; mi < 2; ++mi) {
        /* A frag: row = lane%16, K runs of 8 at (lane/16)*8 and +16 */
        const _Float16* p = Asc + (wm * 32 + mi * 16 + (lane & 15)) * LDS_STRIDE
                                + koff + ((lane >> 4) << 3);
        af[mi] = cat8(*(const v8h*)p, *(const v8h*)(p + 16));
      }
#pragma unroll
      for (int ni = 0; ni < 4; ++ni) {
        /* B frag: col = lane%16, K runs of 8 at (lane/16)*8 and +16 */
        const _Float16* p = Bsc + (wn * 64 + ni * 16 + (lane & 15)) * LDS_STRIDE
                                + koff + ((lane >> 4) << 3);
        bf[ni] = cat8(*(const v8h*)p, *(const v8h*)(p + 16));
      }
#pragma unroll
      for (int mi = 0; mi < 2; ++mi)
#pragma unroll
        for (int ni = 0; ni < 4; ++ni)
          acc[mi][ni] = __builtin_amdgcn_wmma_f32_16x16x32_f16(
              false, af[mi], false, bf[ni], (short)0, acc[mi][ni], false, false);
    }
  }
}

/* C/D element mapping: row = base + v + (lane>=16)*8, col = base + lane%16 */

/* ---- GEMM + QKV scatter epilogue. z=0:Q, z=1:K (t-major), z=2:V (d-major/transposed) */
__global__ __launch_bounds__(256) void gemm_qkv(const _Float16* __restrict__ A,
    const _Float16* __restrict__ Wqt, const _Float16* __restrict__ Wkt,
    const _Float16* __restrict__ Wvt,
    _Float16* __restrict__ q16, _Float16* __restrict__ k16, _Float16* __restrict__ vT16) {
  __shared__ _Float16 As[2][BM * LDS_STRIDE];
  __shared__ _Float16 Bs[2][BN * LDS_STRIDE];
  const int z = blockIdx.z;
  const _Float16* Bt = (z == 0) ? Wqt : ((z == 1) ? Wkt : Wvt);
  v8f acc[2][4];
#pragma unroll
  for (int mi = 0; mi < 2; ++mi)
#pragma unroll
    for (int ni = 0; ni < 4; ++ni) { v8f zf = {}; acc[mi][ni] = zf; }
  const int bm = blockIdx.y * BM, bn = blockIdx.x * BN;
  gemm_core(A, Bt, D_MODEL, As, Bs, bm, bn, acc);

  const int lane = threadIdx.x & 31, wid = threadIdx.x >> 5;
  const int wm = wid & 3, wn = wid >> 2;
#pragma unroll
  for (int mi = 0; mi < 2; ++mi)
#pragma unroll
    for (int ni = 0; ni < 4; ++ni) {
      const int rbase = bm + wm * 32 + mi * 16 + ((lane >> 4) << 3);
      const int c     = bn + wn * 64 + ni * 16 + (lane & 15);
      const int h = c >> 6, d = c & 63;
#pragma unroll
      for (int v = 0; v < 8; ++v) {
        const int r = rbase + v;
        const int b = r >> 11, t = r & (TSEQ - 1);
        const _Float16 val = (_Float16)acc[mi][ni][v];
        if (z == 2)
          vT16[(((size_t)(b * NH + h)) * HD + d) * TSEQ + t] = val;
        else if (z == 1)
          k16[(((size_t)(b * NH + h)) * TSEQ + t) * HD + d] = val;
        else
          q16[(((size_t)(b * NH + h)) * TSEQ + t) * HD + d] = val;
      }
    }
}

/* ---- GEMM + bias + residual, f32 out ---- */
__global__ __launch_bounds__(256) void gemm_bias_res(const _Float16* __restrict__ A,
    const _Float16* __restrict__ Bt, const float* __restrict__ bias,
    const float* __restrict__ res, float* __restrict__ out, int N, int K) {
  __shared__ _Float16 As[2][BM * LDS_STRIDE];
  __shared__ _Float16 Bs[2][BN * LDS_STRIDE];
  v8f acc[2][4];
#pragma unroll
  for (int mi = 0; mi < 2; ++mi)
#pragma unroll
    for (int ni = 0; ni < 4; ++ni) { v8f zf = {}; acc[mi][ni] = zf; }
  const int bm = blockIdx.y * BM, bn = blockIdx.x * BN;
  gemm_core(A, Bt, K, As, Bs, bm, bn, acc);

  const int lane = threadIdx.x & 31, wid = threadIdx.x >> 5;
  const int wm = wid & 3, wn = wid >> 2;
#pragma unroll
  for (int mi = 0; mi < 2; ++mi)
#pragma unroll
    for (int ni = 0; ni < 4; ++ni) {
      const int rbase = bm + wm * 32 + mi * 16 + ((lane >> 4) << 3);
      const int c     = bn + wn * 64 + ni * 16 + (lane & 15);
      const float bv = bias[c];
#pragma unroll
      for (int v = 0; v < 8; ++v) {
        const size_t idx = (size_t)(rbase + v) * N + c;
        out[idx] = acc[mi][ni][v] + bv + res[idx];
      }
    }
}

/* ---- GEMM + bias + exact GELU, f16 out ---- */
__global__ __launch_bounds__(256) void gemm_bias_gelu(const _Float16* __restrict__ A,
    const _Float16* __restrict__ Bt, const float* __restrict__ bias,
    _Float16* __restrict__ out, int N, int K) {
  __shared__ _Float16 As[2][BM * LDS_STRIDE];
  __shared__ _Float16 Bs[2][BN * LDS_STRIDE];
  v8f acc[2][4];
#pragma unroll
  for (int mi = 0; mi < 2; ++mi)
#pragma unroll
    for (int ni = 0; ni < 4; ++ni) { v8f zf = {}; acc[mi][ni] = zf; }
  const int bm = blockIdx.y * BM, bn = blockIdx.x * BN;
  gemm_core(A, Bt, K, As, Bs, bm, bn, acc);

  const int lane = threadIdx.x & 31, wid = threadIdx.x >> 5;
  const int wm = wid & 3, wn = wid >> 2;
#pragma unroll
  for (int mi = 0; mi < 2; ++mi)
#pragma unroll
    for (int ni = 0; ni < 4; ++ni) {
      const int rbase = bm + wm * 32 + mi * 16 + ((lane >> 4) << 3);
      const int c     = bn + wn * 64 + ni * 16 + (lane & 15);
      const float bv = bias[c];
#pragma unroll
      for (int v = 0; v < 8; ++v) {
        float u = acc[mi][ni][v] + bv;
        float g = 0.5f * u * (1.0f + erff(u * 0.70710678118f));
        out[(size_t)(rbase + v) * N + c] = (_Float16)g;
      }
    }
}

/* ---------------- Longformer flash attention: 1 wave per 16-query tile ---------------- */
__device__ __forceinline__ float rowmax16(float x) {
  x = fmaxf(x, __shfl_xor(x, 1, 32));
  x = fmaxf(x, __shfl_xor(x, 2, 32));
  x = fmaxf(x, __shfl_xor(x, 4, 32));
  x = fmaxf(x, __shfl_xor(x, 8, 32));
  return x;
}
__device__ __forceinline__ float rowsum16(float x) {
  x += __shfl_xor(x, 1, 32);
  x += __shfl_xor(x, 2, 32);
  x += __shfl_xor(x, 4, 32);
  x += __shfl_xor(x, 8, 32);
  return x;
}
__device__ __forceinline__ bool allowedQJ(int q, int j) {
  return (j <= q) && ((j + 128 >= q) || (j == 0) || (q == TSEQ - 1));
}

__global__ __launch_bounds__(32) void attn_kernel(const _Float16* __restrict__ q16,
    const _Float16* __restrict__ k16, const _Float16* __restrict__ vT16,
    _Float16* __restrict__ ao16) {
  __shared__ _Float16 Pl[16][32];     /* C-layout -> A-layout remap tile for P */
  const int lane  = threadIdx.x;
  const int bh    = blockIdx.y;
  const int b     = bh >> 4, h = bh & 15;
  const int qBase = blockIdx.x * 16;
  const _Float16* qp = q16 + (size_t)bh * TSEQ * HD;
  const _Float16* kp = k16 + (size_t)bh * TSEQ * HD;
  const _Float16* vh = vT16 + (size_t)bh * HD * TSEQ;

  /* Q fragments (16x64 = two 16x32 A-frags) */
  v16h aQ[2];
#pragma unroll
  for (int c = 0; c < 2; ++c) {
    const _Float16* p = qp + (size_t)(qBase + (lane & 15)) * HD + c * 32 + ((lane >> 4) << 3);
    aQ[c] = cat8(*(const v8h*)p, *(const v8h*)(p + 16));
  }

  v8f o[4];
#pragma unroll
  for (int c = 0; c < 4; ++c) { v8f zf = {}; o[c] = zf; }
  float mrow[8], lrow[8];
#pragma unroll
  for (int v = 0; v < 8; ++v) { mrow[v] = -1.0e30f; lrow[v] = 0.0f; }

  auto proc = [&](int jb) {
    /* scores: two 16x16 tiles vs keys [jb, jb+32), each hd=64 -> 2 chained wmma */
    v8f s0 = {}, s1 = {};
#pragma unroll
    for (int c = 0; c < 2; ++c) {
      {
        const int j = jb + (lane & 15);
        const _Float16* p = kp + (size_t)j * HD + c * 32 + ((lane >> 4) << 3);
        v16h bk = cat8(*(const v8h*)p, *(const v8h*)(p + 16));
        s0 = __builtin_amdgcn_wmma_f32_16x16x32_f16(false, aQ[c], false, bk, (short)0, s0, false, false);
      }
      {
        const int j = jb + 16 + (lane & 15);
        const _Float16* p = kp + (size_t)j * HD + c * 32 + ((lane >> 4) << 3);
        v16h bk = cat8(*(const v8h*)p, *(const v8h*)(p + 16));
        s1 = __builtin_amdgcn_wmma_f32_16x16x32_f16(false, aQ[c], false, bk, (short)0, s1, false, false);
      }
    }
    /* scale + mask + online softmax update */
#pragma unroll
    for (int v = 0; v < 8; ++v) {
      const int q  = qBase + v + ((lane >> 4) << 3);
      const int j0 = jb + (lane & 15);
      float sv0 = allowedQJ(q, j0)      ? s0[v] * 0.125f : -3.0e38f;
      float sv1 = allowedQJ(q, j0 + 16) ? s1[v] * 0.125f : -3.0e38f;
      float bmax = rowmax16(fmaxf(sv0, sv1));
      float mt = fmaxf(mrow[v], bmax);
      float al = __expf(mrow[v] - mt);
      float p0 = __expf(sv0 - mt);
      float p1 = __expf(sv1 - mt);
      float rs = rowsum16(p0 + p1);
      lrow[v] = lrow[v] * al + rs;
      mrow[v] = mt;
      o[0][v] *= al; o[1][v] *= al; o[2][v] *= al; o[3][v] *= al;
      const int prow = v + ((lane >> 4) << 3);
      Pl[prow][lane & 15]        = (_Float16)p0;
      Pl[prow][(lane & 15) + 16] = (_Float16)p1;
    }
    __syncthreads();
    /* O += P(16x32) x V(32x64): A-frag from LDS, V B-frags contiguous (d-major V) */
    const _Float16* pp = &Pl[lane & 15][(lane >> 4) << 3];
    v16h aP = cat8(*(const v8h*)pp, *(const v8h*)(pp + 16));
#pragma unroll
    for (int c = 0; c < 4; ++c) {
      const int d = c * 16 + (lane & 15);
      const _Float16* pv = vh + (size_t)d * TSEQ + jb + ((lane >> 4) << 3);
      v16h bv = cat8(*(const v8h*)pv, *(const v8h*)(pv + 16));
      o[c] = __builtin_amdgcn_wmma_f32_16x16x32_f16(false, aP, false, bv, (short)0, o[c], false, false);
    }
    __syncthreads();
  };

  const bool last = (qBase + 16 >= TSEQ);
  int lo = qBase - 128; if (lo < 0) lo = 0;
  const int kb0   = last ? 0 : (lo >> 5);
  const int kbEnd = last ? (TSEQ >> 5) : (((qBase + 15) >> 5) + 1);
  if (!last && kb0 > 0) proc(0);               /* global column j==0 */
  for (int kb = kb0; kb < kbEnd; ++kb) proc(kb << 5);

  /* normalize + store to [B][T][D] f16 for the Wo GEMM */
#pragma unroll
  for (int v = 0; v < 8; ++v) {
    const float inv = 1.0f / lrow[v];
    const int t = qBase + v + ((lane >> 4) << 3);
#pragma unroll
    for (int c = 0; c < 4; ++c) {
      const int d = c * 16 + (lane & 15);
      ao16[((size_t)(b * TSEQ) + t) * D_MODEL + h * HD + d] = (_Float16)(o[c][v] * inv);
    }
  }
}

/* ---------------- LayerNorm: one block per row, f32 in -> f16 out ---------------- */
__global__ __launch_bounds__(256) void ln_kernel(const float* __restrict__ x,
    const float* __restrict__ g, const float* __restrict__ bta,
    _Float16* __restrict__ out) {
  __shared__ float red[256], red2[256];
  const int row = blockIdx.x, tid = threadIdx.x;
  const float* xr = x + (size_t)row * D_MODEL;
  float vals[4], s = 0.f, s2 = 0.f;
#pragma unroll
  for (int i = 0; i < 4; ++i) {
    float v = xr[tid + i * 256];
    vals[i] = v; s += v; s2 += v * v;
  }
  red[tid] = s; red2[tid] = s2;
  __syncthreads();
  for (int o2 = 128; o2 > 0; o2 >>= 1) {
    if (tid < o2) { red[tid] += red[tid + o2]; red2[tid] += red2[tid + o2]; }
    __syncthreads();
  }
  const float mu  = red[0] * (1.0f / D_MODEL);
  const float var = red2[0] * (1.0f / D_MODEL) - mu * mu;
  const float rstd = rsqrtf(var + 1e-5f);
#pragma unroll
  for (int i = 0; i < 4; ++i) {
    const int c = tid + i * 256;
    out[(size_t)row * D_MODEL + c] = (_Float16)((vals[i] - mu) * rstd * g[c] + bta[c]);
  }
}

/* ---------------- weight transpose + fp32->f16 convert: W[K][N] -> Wt[N][K] -------- */
__global__ __launch_bounds__(256) void transpose_cvt(const float* __restrict__ W,
    _Float16* __restrict__ Wt, int K, int N) {
  __shared__ _Float16 tile[32][33];
  const int n0 = blockIdx.x * 32, k0 = blockIdx.y * 32;
  for (int i = threadIdx.y; i < 32; i += 8)
    tile[i][threadIdx.x] = (_Float16)W[(size_t)(k0 + i) * N + n0 + threadIdx.x];
  __syncthreads();
  for (int i = threadIdx.y; i < 32; i += 8)
    Wt[(size_t)(n0 + i) * K + k0 + threadIdx.x] = tile[threadIdx.x][i];
}

extern "C" void kernel_launch(void* const* d_in, const int* in_sizes, int n_in,
                              void* d_out, int out_size, void* d_ws, size_t ws_size,
                              hipStream_t stream) {
  const float* x    = (const float*)d_in[0];
  const float* ln1g = (const float*)d_in[1];
  const float* ln1b = (const float*)d_in[2];
  const float* ln2g = (const float*)d_in[3];
  const float* ln2b = (const float*)d_in[4];
  const float* Wq   = (const float*)d_in[5];
  const float* Wk   = (const float*)d_in[6];
  const float* Wv   = (const float*)d_in[7];
  const float* Wo   = (const float*)d_in[8];
  const float* bo   = (const float*)d_in[9];
  const float* W1   = (const float*)d_in[10];
  const float* b1   = (const float*)d_in[11];
  const float* W2   = (const float*)d_in[12];
  const float* b2   = (const float*)d_in[13];
  float* out = (float*)d_out;

  /* workspace carve-up */
  char* w = (char*)d_ws;
  _Float16* h16  = (_Float16*)w; w += (size_t)MROWS * D_MODEL * 2;
  _Float16* q16  = (_Float16*)w; w += (size_t)MROWS * D_MODEL * 2;
  _Float16* k16  = (_Float16*)w; w += (size_t)MROWS * D_MODEL * 2;
  _Float16* vT16 = (_Float16*)w; w += (size_t)MROWS * D_MODEL * 2;
  _Float16* ao16 = (_Float16*)w; w += (size_t)MROWS * D_MODEL * 2;
  _Float16* h216 = (_Float16*)w; w += (size_t)MROWS * D_MODEL * 2;
  _Float16* mid16= (_Float16*)w; w += (size_t)MROWS * DFF * 2;
  float*    x1   = (float*)w;    w += (size_t)MROWS * D_MODEL * 4;
  _Float16* wqt  = (_Float16*)w; w += (size_t)D_MODEL * D_MODEL * 2;
  _Float16* wkt  = (_Float16*)w; w += (size_t)D_MODEL * D_MODEL * 2;
  _Float16* wvt  = (_Float16*)w; w += (size_t)D_MODEL * D_MODEL * 2;
  _Float16* wot  = (_Float16*)w; w += (size_t)D_MODEL * D_MODEL * 2;
  _Float16* w1t  = (_Float16*)w; w += (size_t)D_MODEL * DFF * 2;
  _Float16* w2t  = (_Float16*)w; w += (size_t)DFF * D_MODEL * 2;

  dim3 tb(32, 8);
  transpose_cvt<<<dim3(D_MODEL / 32, D_MODEL / 32), tb, 0, stream>>>(Wq, wqt, D_MODEL, D_MODEL);
  transpose_cvt<<<dim3(D_MODEL / 32, D_MODEL / 32), tb, 0, stream>>>(Wk, wkt, D_MODEL, D_MODEL);
  transpose_cvt<<<dim3(D_MODEL / 32, D_MODEL / 32), tb, 0, stream>>>(Wv, wvt, D_MODEL, D_MODEL);
  transpose_cvt<<<dim3(D_MODEL / 32, D_MODEL / 32), tb, 0, stream>>>(Wo, wot, D_MODEL, D_MODEL);
  transpose_cvt<<<dim3(DFF / 32, D_MODEL / 32), tb, 0, stream>>>(W1, w1t, D_MODEL, DFF);
  transpose_cvt<<<dim3(D_MODEL / 32, DFF / 32), tb, 0, stream>>>(W2, w2t, DFF, D_MODEL);

  ln_kernel<<<MROWS, 256, 0, stream>>>(x, ln1g, ln1b, h16);

  gemm_qkv<<<dim3(D_MODEL / BN, MROWS / BM, 3), 256, 0, stream>>>(
      h16, wqt, wkt, wvt, q16, k16, vT16);

  attn_kernel<<<dim3(TSEQ / 16, BATCH * NH), 32, 0, stream>>>(q16, k16, vT16, ao16);

  gemm_bias_res<<<dim3(D_MODEL / BN, MROWS / BM), 256, 0, stream>>>(
      ao16, wot, bo, x, x1, D_MODEL, D_MODEL);

  ln_kernel<<<MROWS, 256, 0, stream>>>(x1, ln2g, ln2b, h216);

  gemm_bias_gelu<<<dim3(DFF / BN, MROWS / BM), 256, 0, stream>>>(
      h216, w1t, b1, mid16, DFF, D_MODEL);

  gemm_bias_res<<<dim3(D_MODEL / BN, MROWS / BM), 256, 0, stream>>>(
      mid16, w2t, b2, x1, out, D_MODEL, DFF);
}
